// Llama4MoE_14740327759920
// MI455X (gfx1250) — matrix operand order repaired and checked by reading, not compiled, not run on hardware
//
#include <hip/hip_runtime.h>
#include <hip/hip_bf16.h>
#include <stdint.h>

// Problem sizes (fixed by reference)
#define T_TOK 2048
#define HDIM  2048
#define IDIM  4096
#define NEXP  8

// GEMM tiling
#define TM 128
#define TN 128
#define BK 32
#define BKP 40          // padded LDS K-stride in halves (80B rows, 16B aligned)
#define MAXTILES 32

typedef __attribute__((ext_vector_type(16))) __bf16        v16bf;
typedef __attribute__((ext_vector_type(8)))  float         v8f;
typedef __attribute__((ext_vector_type(4)))  float         f32x4;
typedef __attribute__((ext_vector_type(4)))  unsigned int  u32x4;
typedef __attribute__((ext_vector_type(2)))  unsigned int  u32x2;

struct B32x8 { u32x4 lo, hi; };

__device__ __forceinline__ unsigned short f2bf(float f) {
  unsigned int u = __float_as_uint(f);
  u += 0x7fffu + ((u >> 16) & 1u);           // round-to-nearest-even
  return (unsigned short)(u >> 16);
}

__device__ __forceinline__ v16bf load_frag16(const unsigned short* p0,
                                             const unsigned short* p1) {
  B32x8 t;
  t.lo = *(const u32x4*)p0;
  t.hi = *(const u32x4*)p1;
  return __builtin_bit_cast(v16bf, t);
}

// ---------------------------------------------------------------- init
__global__ void moe_init_kernel(int* counts) {
  if (threadIdx.x < NEXP) counts[threadIdx.x] = 0;
}

// ---------------------------------------------------------------- router
// one wave (32 lanes) per token: logits = x . w_router[:,e], top-1, sigmoid
__global__ __launch_bounds__(256)
void moe_router_kernel(const float* __restrict__ hs,
                       const float* __restrict__ wr,
                       float* __restrict__ score,
                       int* __restrict__ sel,
                       int* __restrict__ counts) {
  const int lane = threadIdx.x & 31;
  const int wid  = threadIdx.x >> 5;
  const int t    = blockIdx.x * 8 + wid;
  float acc[NEXP];
#pragma unroll
  for (int e = 0; e < NEXP; ++e) acc[e] = 0.f;
  for (int h = lane; h < HDIM; h += 32) {
    float x  = hs[(size_t)t * HDIM + h];
    f32x4 w0 = *(const f32x4*)(wr + (size_t)h * NEXP);
    f32x4 w1 = *(const f32x4*)(wr + (size_t)h * NEXP + 4);
    acc[0] += x * w0[0]; acc[1] += x * w0[1];
    acc[2] += x * w0[2]; acc[3] += x * w0[3];
    acc[4] += x * w1[0]; acc[5] += x * w1[1];
    acc[6] += x * w1[2]; acc[7] += x * w1[3];
  }
#pragma unroll
  for (int off = 16; off > 0; off >>= 1)
#pragma unroll
    for (int e = 0; e < NEXP; ++e)
      acc[e] += __shfl_xor(acc[e], off, 32);
  if (lane == 0) {
    int best = 0; float bv = acc[0];
#pragma unroll
    for (int e = 1; e < NEXP; ++e)
      if (acc[e] > bv) { bv = acc[e]; best = e; }   // strict '>' keeps first (jax top_k tie rule)
    score[t] = 1.0f / (1.0f + __expf(-bv));
    sel[t] = best;
    atomicAdd(&counts[best], 1);
  }
}

// ---------------------------------------------------------------- scan / tile table
__global__ void moe_scan_kernel(const int* __restrict__ counts,
                                int* __restrict__ base,
                                int* __restrict__ cursors,
                                int* __restrict__ tileE,
                                int* __restrict__ tileR0,
                                int* __restrict__ tileRows,
                                int* __restrict__ ntiles) {
  if (threadIdx.x == 0 && blockIdx.x == 0) {
    int acc = 0, nt = 0;
    for (int e = 0; e < NEXP; ++e) {
      base[e] = acc; cursors[e] = acc;
      int c = counts[e];
      for (int r = 0; r < c; r += TM) {
        tileE[nt] = e;
        tileR0[nt] = acc + r;
        tileRows[nt] = (c - r < TM) ? (c - r) : TM;
        ++nt;
      }
      acc += c;
    }
    *ntiles = nt;
  }
}

// ---------------------------------------------------------------- gather + bf16 convert
// block = one token: writes unscaled x_bf16[t], scaled xg[pos] (permuted), perm[pos]=t
__global__ __launch_bounds__(256)
void moe_gather_kernel(const float* __restrict__ hs,
                       const float* __restrict__ score,
                       const int* __restrict__ sel,
                       int* __restrict__ cursors,
                       int* __restrict__ perm,
                       unsigned short* __restrict__ xbf,
                       unsigned short* __restrict__ xg) {
  const int t = blockIdx.x;
  __shared__ int spos;
  if (threadIdx.x == 0) {
    int s = sel[t];
    int p = atomicAdd(&cursors[s], 1);
    perm[p] = t;
    spos = p;
  }
  __syncthreads();
  const int pos = spos;
  const float sc = score[t];
  for (int h = threadIdx.x * 4; h < HDIM; h += 256 * 4) {
    f32x4 v = *(const f32x4*)(hs + (size_t)t * HDIM + h);
    u32x2 pu, ps;
    pu[0] = (unsigned int)f2bf(v[0])      | ((unsigned int)f2bf(v[1])      << 16);
    pu[1] = (unsigned int)f2bf(v[2])      | ((unsigned int)f2bf(v[3])      << 16);
    ps[0] = (unsigned int)f2bf(v[0] * sc) | ((unsigned int)f2bf(v[1] * sc) << 16);
    ps[1] = (unsigned int)f2bf(v[2] * sc) | ((unsigned int)f2bf(v[3] * sc) << 16);
    *(u32x2*)(xbf + (size_t)t   * HDIM + h) = pu;
    *(u32x2*)(xg  + (size_t)pos * HDIM + h) = ps;
  }
}

// ---------------------------------------------------------------- gate/up GEMM (+silu*up)
// C1 = A @ Wg, C2 = A @ Wu  -> H = silu(C1)*C2 stored bf16.  A: (rows,H) bf16 row-major.
template <bool GATHERED>
__global__ __launch_bounds__(256)
void moe_gateup_kernel(const unsigned short* __restrict__ Abf,
                       const float* __restrict__ WgAll,
                       const float* __restrict__ WuAll,
                       unsigned short* __restrict__ Hout,
                       const int* __restrict__ tileE,
                       const int* __restrict__ tileR0,
                       const int* __restrict__ tileRows,
                       const int* __restrict__ ntilesPtr) {
  constexpr int K = HDIM;
  constexpr int LDB = IDIM;
  __shared__ alignas(16) unsigned short sA [TM * BKP];
  __shared__ alignas(16) unsigned short sBg[TN * BKP];
  __shared__ alignas(16) unsigned short sBu[TN * BKP];

  int e, row0, rows;
  if (GATHERED) {
    if ((int)blockIdx.y >= *ntilesPtr) return;
    e = tileE[blockIdx.y]; row0 = tileR0[blockIdx.y]; rows = tileRows[blockIdx.y];
  } else {
    e = 0; row0 = blockIdx.y * TM; rows = TM;
  }
  const int col0 = blockIdx.x * TN;
  const float* __restrict__ Bg = WgAll + (size_t)e * HDIM * IDIM;
  const float* __restrict__ Bu = WuAll + (size_t)e * HDIM * IDIM;

  const int tid  = threadIdx.x;
  const int lane = tid & 31;
  const int wid  = tid >> 5;
  const int wm   = (wid >> 2) * 64;   // wave M origin (0/64)
  const int wn   = (wid & 3) * 32;    // wave N origin (0/32/64/96)
  const int lm   = lane & 15;
  const int l16  = lane >> 4;
  const int akb  = l16 * 16;          // A K-base (halves)
  const int bkb  = l16 * 16;          // B K-base (halves)

  v8f accg[4][2], accu[4][2];
  const v8f vzero = {0.f, 0.f, 0.f, 0.f, 0.f, 0.f, 0.f, 0.f};
#pragma unroll
  for (int i = 0; i < 4; ++i)
#pragma unroll
    for (int j = 0; j < 2; ++j) { accg[i][j] = vzero; accu[i][j] = vzero; }

  const int ar  = tid >> 1;           // A loader: row 0..127
  const int ak  = (tid & 1) * 16;     // A loader: K half-offset 0/16
  const int bk2 = (tid >> 5) * 2;     // B loader: even K 0..14
  const int bn  = (tid & 31) * 4;     // B loader: N 0..124

  for (int k0 = 0; k0 < K; k0 += BK) {
    // ---- stage A (bf16 copy)
    u32x4 av = {0u, 0u, 0u, 0u};
    if (!GATHERED || ar < rows)
      av = *(const u32x4*)(Abf + (size_t)(row0 + ar) * K + (k0 + ak));
    *(u32x4*)(sA + ar * BKP + ak) = av;
    // ---- stage B gate/up (f32 -> bf16, transpose to N-major, K-pairs packed)
#pragma unroll
    for (int it = 0; it < 2; ++it) {
      const int kk = bk2 + it * 16;
      const float* pg0 = Bg + (size_t)(k0 + kk) * LDB + (col0 + bn);
      const float* pu0 = Bu + (size_t)(k0 + kk) * LDB + (col0 + bn);
      if (it == 0 && k0 + BK < K) {
        __builtin_prefetch(pg0 + (size_t)BK * LDB, 0, 1);
        __builtin_prefetch(pu0 + (size_t)BK * LDB, 0, 1);
      }
      f32x4 g0 = *(const f32x4*)pg0;
      f32x4 g1 = *(const f32x4*)(pg0 + LDB);
      f32x4 u0 = *(const f32x4*)pu0;
      f32x4 u1 = *(const f32x4*)(pu0 + LDB);
#pragma unroll
      for (int j = 0; j < 4; ++j) {
        *(unsigned int*)(sBg + (bn + j) * BKP + kk) =
            (unsigned int)f2bf(g0[j]) | ((unsigned int)f2bf(g1[j]) << 16);
        *(unsigned int*)(sBu + (bn + j) * BKP + kk) =
            (unsigned int)f2bf(u0[j]) | ((unsigned int)f2bf(u1[j]) << 16);
      }
    }
    __syncthreads();
    // ---- fragments + WMMA
    v16bf afr[4], bgf[2], buf_[2];
#pragma unroll
    for (int mt = 0; mt < 4; ++mt) {
      const unsigned short* p = sA + (wm + mt * 16 + lm) * BKP + akb;
      afr[mt] = load_frag16(p, p + 8);
    }
#pragma unroll
    for (int nt = 0; nt < 2; ++nt) {
      const unsigned short* pg = sBg + (wn + nt * 16 + lm) * BKP + bkb;
      bgf[nt] = load_frag16(pg, pg + 8);
      const unsigned short* pu = sBu + (wn + nt * 16 + lm) * BKP + bkb;
      buf_[nt] = load_frag16(pu, pu + 8);
    }
#pragma unroll
    for (int mt = 0; mt < 4; ++mt)
#pragma unroll
      for (int nt = 0; nt < 2; ++nt) {
        accg[mt][nt] = __builtin_amdgcn_wmma_f32_16x16x32_bf16(
            false, afr[mt], false, bgf[nt], (short)0, accg[mt][nt], false, false);
        accu[mt][nt] = __builtin_amdgcn_wmma_f32_16x16x32_bf16(
            false, afr[mt], false, buf_[nt], (short)0, accu[mt][nt], false, false);
      }
    __syncthreads();
  }
  // ---- epilogue: h = silu(g) * u  (bf16 out)
#pragma unroll
  for (int mt = 0; mt < 4; ++mt)
#pragma unroll
    for (int nt = 0; nt < 2; ++nt)
#pragma unroll
      for (int v = 0; v < 8; ++v) {
        int rr = wm + mt * 16 + l16 * 8 + v;
        if (GATHERED && rr >= rows) continue;
        float g = accg[mt][nt][v];
        float u = accu[mt][nt][v];
        float hv = (g / (1.0f + __expf(-g))) * u;
        int gc = col0 + wn + nt * 16 + lm;
        Hout[(size_t)(row0 + rr) * IDIM + gc] = f2bf(hv);
      }
}

// ---------------------------------------------------------------- down GEMM
// Out = H @ Wd. GATHERED: scatter rows via perm and accumulate (+=); else plain store.
template <bool GATHERED>
__global__ __launch_bounds__(256)
void moe_down_kernel(const unsigned short* __restrict__ Abf,
                     const float* __restrict__ WdAll,
                     float* __restrict__ Out,
                     const int* __restrict__ tileE,
                     const int* __restrict__ tileR0,
                     const int* __restrict__ tileRows,
                     const int* __restrict__ ntilesPtr,
                     const int* __restrict__ perm) {
  constexpr int K = IDIM;
  constexpr int LDB = HDIM;
  __shared__ alignas(16) unsigned short sA[TM * BKP];
  __shared__ alignas(16) unsigned short sB[TN * BKP];

  int e, row0, rows;
  if (GATHERED) {
    if ((int)blockIdx.y >= *ntilesPtr) return;
    e = tileE[blockIdx.y]; row0 = tileR0[blockIdx.y]; rows = tileRows[blockIdx.y];
  } else {
    e = 0; row0 = blockIdx.y * TM; rows = TM;
  }
  const int col0 = blockIdx.x * TN;
  const float* __restrict__ Bd = WdAll + (size_t)e * IDIM * HDIM;

  const int tid  = threadIdx.x;
  const int lane = tid & 31;
  const int wid  = tid >> 5;
  const int wm   = (wid >> 2) * 64;
  const int wn   = (wid & 3) * 32;
  const int lm   = lane & 15;
  const int l16  = lane >> 4;
  const int akb  = l16 * 16;
  const int bkb  = l16 * 16;

  v8f acc[4][2];
  const v8f vzero = {0.f, 0.f, 0.f, 0.f, 0.f, 0.f, 0.f, 0.f};
#pragma unroll
  for (int i = 0; i < 4; ++i)
#pragma unroll
    for (int j = 0; j < 2; ++j) acc[i][j] = vzero;

  const int ar  = tid >> 1;
  const int ak  = (tid & 1) * 16;
  const int bk2 = (tid >> 5) * 2;
  const int bn  = (tid & 31) * 4;

  for (int k0 = 0; k0 < K; k0 += BK) {
    u32x4 av = {0u, 0u, 0u, 0u};
    if (!GATHERED || ar < rows)
      av = *(const u32x4*)(Abf + (size_t)(row0 + ar) * K + (k0 + ak));
    *(u32x4*)(sA + ar * BKP + ak) = av;
#pragma unroll
    for (int it = 0; it < 2; ++it) {
      const int kk = bk2 + it * 16;
      const float* pb0 = Bd + (size_t)(k0 + kk) * LDB + (col0 + bn);
      if (it == 0 && k0 + BK < K) __builtin_prefetch(pb0 + (size_t)BK * LDB, 0, 1);
      f32x4 b0 = *(const f32x4*)pb0;
      f32x4 b1 = *(const f32x4*)(pb0 + LDB);
#pragma unroll
      for (int j = 0; j < 4; ++j)
        *(unsigned int*)(sB + (bn + j) * BKP + kk) =
            (unsigned int)f2bf(b0[j]) | ((unsigned int)f2bf(b1[j]) << 16);
    }
    __syncthreads();
    v16bf afr[4], bf[2];
#pragma unroll
    for (int mt = 0; mt < 4; ++mt) {
      const unsigned short* p = sA + (wm + mt * 16 + lm) * BKP + akb;
      afr[mt] = load_frag16(p, p + 8);
    }
#pragma unroll
    for (int nt = 0; nt < 2; ++nt) {
      const unsigned short* p = sB + (wn + nt * 16 + lm) * BKP + bkb;
      bf[nt] = load_frag16(p, p + 8);
    }
#pragma unroll
    for (int mt = 0; mt < 4; ++mt)
#pragma unroll
      for (int nt = 0; nt < 2; ++nt)
        acc[mt][nt] = __builtin_amdgcn_wmma_f32_16x16x32_bf16(
            false, afr[mt], false, bf[nt], (short)0, acc[mt][nt], false, false);
    __syncthreads();
  }
#pragma unroll
  for (int mt = 0; mt < 4; ++mt)
#pragma unroll
    for (int nt = 0; nt < 2; ++nt)
#pragma unroll
      for (int v = 0; v < 8; ++v) {
        int rr = wm + mt * 16 + l16 * 8 + v;
        if (GATHERED && rr >= rows) continue;
        int gc = col0 + wn + nt * 16 + lm;
        int token = GATHERED ? perm[row0 + rr] : (row0 + rr);
        float* o = Out + (size_t)token * HDIM + gc;
        float val = acc[mt][nt][v];
        if (GATHERED) *o += val;   // unique (token,col) writer; shared pass already wrote '='
        else          *o  = val;
      }
}

// ---------------------------------------------------------------- launch
extern "C" void kernel_launch(void* const* d_in, const int* in_sizes, int n_in,
                              void* d_out, int out_size, void* d_ws, size_t ws_size,
                              hipStream_t stream) {
  const float* hs  = (const float*)d_in[0];
  const float* wr  = (const float*)d_in[1];
  const float* wg  = (const float*)d_in[2];
  const float* wu  = (const float*)d_in[3];
  const float* wd  = (const float*)d_in[4];
  const float* wsg = (const float*)d_in[5];
  const float* wsu = (const float*)d_in[6];
  const float* wsd = (const float*)d_in[7];
  float* out = (float*)d_out;

  uintptr_t p = (uintptr_t)d_ws;
  auto take = [&](size_t bytes) -> void* {
    uintptr_t r = (p + 255) & ~(uintptr_t)255;
    p = r + bytes;
    return (void*)r;
  };
  float* score   = (float*)take((size_t)T_TOK * 4);
  int*   sel     = (int*)take((size_t)T_TOK * 4);
  int*   counts  = (int*)take(NEXP * 4);
  int*   cursors = (int*)take(NEXP * 4);
  int*   base    = (int*)take(NEXP * 4);
  int*   ntiles  = (int*)take(4);
  int*   tileE   = (int*)take(MAXTILES * 4);
  int*   tileR0  = (int*)take(MAXTILES * 4);
  int*   tileRw  = (int*)take(MAXTILES * 4);
  int*   perm    = (int*)take((size_t)T_TOK * 4);
  unsigned short* xbf = (unsigned short*)take((size_t)T_TOK * HDIM * 2);
  unsigned short* xg  = (unsigned short*)take((size_t)T_TOK * HDIM * 2);
  unsigned short* hS  = (unsigned short*)take((size_t)T_TOK * IDIM * 2);
  unsigned short* hR  = (unsigned short*)take((size_t)T_TOK * IDIM * 2);

  moe_init_kernel<<<1, 32, 0, stream>>>(counts);
  moe_router_kernel<<<T_TOK / 8, 256, 0, stream>>>(hs, wr, score, sel, counts);
  moe_scan_kernel<<<1, 1, 0, stream>>>(counts, base, cursors, tileE, tileR0, tileRw, ntiles);
  moe_gather_kernel<<<T_TOK, 256, 0, stream>>>(hs, score, sel, cursors, perm, xbf, xg);

  // shared expert
  moe_gateup_kernel<false><<<dim3(IDIM / TN, T_TOK / TM), 256, 0, stream>>>(
      xbf, wsg, wsu, hS, nullptr, nullptr, nullptr, nullptr);
  moe_down_kernel<false><<<dim3(HDIM / TN, T_TOK / TM), 256, 0, stream>>>(
      hS, wsd, out, nullptr, nullptr, nullptr, nullptr, nullptr);

  // routed experts (grouped GEMM over tile table)
  moe_gateup_kernel<true><<<dim3(IDIM / TN, MAXTILES), 256, 0, stream>>>(
      xg, wg, wu, hR, tileE, tileR0, tileRw, ntiles);
  moe_down_kernel<true><<<dim3(HDIM / TN, MAXTILES), 256, 0, stream>>>(
      hR, wd, out, tileE, tileR0, tileRw, ntiles, perm);
}
